// final_model_52561809768736
// MI455X (gfx1250) — matrix operand order repaired and checked by reading, not compiled, not run on hardware
//
#include <hip/hip_runtime.h>
#include <math.h>

#define N_NODES 20000
#define E_EDGES 640000
#define L_LAYERS 2
#define DIN 1546
#define DH 128
#define DFF 512
#define EPS_LN 1e-5f
#define SLOPE 0.01f

// padded dims
#define MP 20096          // 157 * 128 rows
#define KP_IN 1568        // 49 * 32, padded DIN for K dimension
#define NCP_LIN 1664      // 13 * 128, allocated rows for linW^T (B of 128->1546 GEMM)

typedef __bf16 bf16_t;
typedef bf16_t v8bf  __attribute__((ext_vector_type(8)));
typedef bf16_t v16bf __attribute__((ext_vector_type(16)));
typedef float  v8f   __attribute__((ext_vector_type(8)));
typedef unsigned int u32x4 __attribute__((ext_vector_type(4)));
typedef int i32x4 __attribute__((ext_vector_type(4)));
typedef int i32x8 __attribute__((ext_vector_type(8)));

#if defined(__has_builtin)
#if __has_builtin(__builtin_amdgcn_tensor_load_to_lds) && __has_builtin(__builtin_amdgcn_s_wait_tensorcnt)
#define HAVE_TDM 1
#endif
#endif

__device__ __forceinline__ bf16_t f2bf(float f) {
  unsigned u = __builtin_bit_cast(unsigned, f);
  unsigned r = u + 0x7FFFu + ((u >> 16) & 1u);   // round-to-nearest-even
  unsigned short h = (unsigned short)(r >> 16);
  return __builtin_bit_cast(bf16_t, h);
}

__device__ __forceinline__ v16bf cat16(v8bf lo, v8bf hi) {
  return __builtin_shufflevector(lo, hi, 0,1,2,3,4,5,6,7,8,9,10,11,12,13,14,15);
}

#define BM 128
#define BN 128
#define BK 32
#define LDT 40   // padded LDS row stride in bf16 (80B = 64B row + 16B pad)

#ifdef HAVE_TDM
// Issue one TDM 2D tile load: 128 rows x 32 bf16 from gptr (row stride =
// stride_elems bf16) into LDS at lds_byte_addr, with LDS padding of 4 DWORDs
// after every 16 DWORDs stored (=> LDS row stride LDT=40 bf16).
__device__ __forceinline__ void tdm_load_tile(const bf16_t* gptr, int stride_elems,
                                              unsigned lds_byte_addr)
{
  unsigned long long ga = (unsigned long long)(uintptr_t)gptr;
  u32x4 g0;
  g0[0] = 1u;                                          // count=1, user mode
  g0[1] = lds_byte_addr;                               // lds_addr
  g0[2] = (unsigned)(ga & 0xFFFFFFFFu);                // global_addr[31:0]
  g0[3] = (unsigned)((ga >> 32) & 0x01FFFFFFu)         // global_addr[56:32]
        | (2u << 30);                                  // type = 2 (image)
  i32x8 g1;
  g1[0] = (1 << 16)       // data_size = 2 bytes
        | (1 << 20)       // pad_enable
        | (3 << 22)       // pad_interval: 16 DWORDs between pads
        | (3 << 25);      // pad_amount: 4 DWORDs
  g1[1] = (BK << 16);     // tensor_dim0[15:0] = 32 (in bits 63:48)
  g1[2] = (BM << 16);     // tensor_dim1[15:0] = 128 (bits 95:80)
  g1[3] = (BK << 16);     // tile_dim0 = 32 (bits 127:112)
  g1[4] = BM;             // tile_dim1 = 128 (bits 143:128)
  g1[5] = stride_elems;   // tensor_dim0_stride[31:0]
  g1[6] = 0;
  g1[7] = 0;
  i32x4 z4 = {0, 0, 0, 0};
#if __clang_major__ >= 23
  i32x8 z8 = {0, 0, 0, 0, 0, 0, 0, 0};
  __builtin_amdgcn_tensor_load_to_lds(g0, g1, z4, z4, z8, 0);
#else
  __builtin_amdgcn_tensor_load_to_lds(g0, g1, z4, z4, 0);
#endif
}
#endif

// ---------------------------------------------------------------------------
// C = act(A @ B + bias). A: bf16 [rows x lda] row-major, zero-padded.
// Bw: bf16 [cols x ldb] K-major per output column (pre-transposed weights).
// Kp multiple of 32. Output: f32 (Cf, guarded) or padded bf16 (Cb).
// Block = 256 thr (8 waves), tile 128x128, wave tile 32x64.
// Double-buffered LDS; tiles staged by the Tensor Data Mover (wave 0 posts
// descriptors, s_wait_tensorcnt + workgroup barrier publishes), falling back
// to register prefetch if the TDM builtin is unavailable.
// ---------------------------------------------------------------------------
__global__ __launch_bounds__(256)
void gemm_bf16_wmma(const bf16_t* __restrict__ A, int lda,
                    const bf16_t* __restrict__ Bw, int ldb,
                    const float* __restrict__ bias, int bias_n, int act,
                    float* __restrict__ Cf, bf16_t* __restrict__ Cb, int ldc,
                    int M, int Nc, int Kp)
{
  __shared__ bf16_t As[2][BM * LDT];
  __shared__ bf16_t Bs[2][BN * LDT];

  const int tid  = threadIdx.x;
  const int m0   = blockIdx.x * BM;
  const int n0   = blockIdx.y * BN;
  const int wave = tid >> 5;
  const int lane = tid & 31;
  const int lm   = lane & 15;
  const int half = lane >> 4;
  const int wr   = wave & 3;   // row group: rows 32*wr .. +31
  const int wc   = wave >> 2;  // col group: cols 64*wc .. +63

  const bf16_t* Ab = A  + (long long)m0 * lda;
  const bf16_t* Bb = Bw + (long long)n0 * ldb;

  v8f acc[2][4];
#pragma unroll
  for (int ai = 0; ai < 2; ++ai)
#pragma unroll
    for (int bi = 0; bi < 4; ++bi) acc[ai][bi] = v8f{};

#ifdef HAVE_TDM
  const unsigned ldsA0 = (unsigned)(uintptr_t)&As[0][0];
  const unsigned ldsA1 = (unsigned)(uintptr_t)&As[1][0];
  const unsigned ldsB0 = (unsigned)(uintptr_t)&Bs[0][0];
  const unsigned ldsB1 = (unsigned)(uintptr_t)&Bs[1][0];
  if (wave == 0) {
    tdm_load_tile(Ab, lda, ldsA0);
    tdm_load_tile(Bb, ldb, ldsB0);
    __builtin_amdgcn_s_wait_tensorcnt(0);
  }
  __syncthreads();
#else
  const int r0 = tid >> 2;          // 0..63 (rows r0 and r0+64)
  const int c8 = (tid & 3) * 8;     // 0,8,16,24 within 32-wide K slab
  v8bf a0 = *(const v8bf*)&Ab[(long long)r0 * lda + c8];
  v8bf a1 = *(const v8bf*)&Ab[(long long)(r0 + 64) * lda + c8];
  v8bf b0 = *(const v8bf*)&Bb[(long long)r0 * ldb + c8];
  v8bf b1 = *(const v8bf*)&Bb[(long long)(r0 + 64) * ldb + c8];
  *(v8bf*)&As[0][r0 * LDT + c8]        = a0;
  *(v8bf*)&As[0][(r0 + 64) * LDT + c8] = a1;
  *(v8bf*)&Bs[0][r0 * LDT + c8]        = b0;
  *(v8bf*)&Bs[0][(r0 + 64) * LDT + c8] = b1;
  __syncthreads();
#endif

  int buf = 0;
  for (int k0 = 0; k0 < Kp; k0 += BK) {
    const bool has_next = (k0 + BK) < Kp;

#ifdef HAVE_TDM
    if (has_next && wave == 0) {
      const int kn = k0 + BK;
      tdm_load_tile(Ab + kn, lda, buf ? ldsA0 : ldsA1);
      tdm_load_tile(Bb + kn, ldb, buf ? ldsB0 : ldsB1);
    }
#else
    if (has_next) {
      const int kn = k0 + BK;
      a0 = *(const v8bf*)&Ab[(long long)r0 * lda + kn + c8];
      a1 = *(const v8bf*)&Ab[(long long)(r0 + 64) * lda + kn + c8];
      b0 = *(const v8bf*)&Bb[(long long)r0 * ldb + kn + c8];
      b1 = *(const v8bf*)&Bb[(long long)(r0 + 64) * ldb + kn + c8];
    }
#endif

    // Issue ALL fragment loads first (12x ds_load_b128 into distinct regs),
    // then the 8 WMMAs: staggers s_wait_dscnt and overlaps LDS with compute.
    // A fragments: per-lane K = {half*8+0..7, 16+half*8+0..7}
    v16bf af[2];
#pragma unroll
    for (int ai = 0; ai < 2; ++ai) {
      const bf16_t* arow = &As[buf][(wr * 32 + ai * 16 + lm) * LDT];
      af[ai] = cat16(*(const v8bf*)&arow[half * 8],
                     *(const v8bf*)&arow[16 + half * 8]);
    }
    // B fragments: per-lane K = {half*8+0..7, 16+half*8+0..7}
    v16bf bfr[4];
#pragma unroll
    for (int bi = 0; bi < 4; ++bi) {
      const bf16_t* brow = &Bs[buf][(wc * 64 + bi * 16 + lm) * LDT];
      bfr[bi] = cat16(*(const v8bf*)&brow[half * 8],
                      *(const v8bf*)&brow[16 + half * 8]);
    }
#pragma unroll
    for (int bi = 0; bi < 4; ++bi)
#pragma unroll
      for (int ai = 0; ai < 2; ++ai)
        acc[ai][bi] = __builtin_amdgcn_wmma_f32_16x16x32_bf16(
            false, af[ai], false, bfr[bi], (short)0, acc[ai][bi], false, false);

#ifdef HAVE_TDM
    if (has_next && wave == 0) __builtin_amdgcn_s_wait_tensorcnt(0);
#else
    if (has_next) {
      const int nb = buf ^ 1;
      *(v8bf*)&As[nb][r0 * LDT + c8]        = a0;
      *(v8bf*)&As[nb][(r0 + 64) * LDT + c8] = a1;
      *(v8bf*)&Bs[nb][r0 * LDT + c8]        = b0;
      *(v8bf*)&Bs[nb][(r0 + 64) * LDT + c8] = b1;
    }
#endif
    __syncthreads();
    buf ^= 1;
  }

  // epilogue: C/D layout -> VGPR r holds M = r + 8*half, lane lm holds col
#pragma unroll
  for (int ai = 0; ai < 2; ++ai) {
#pragma unroll
    for (int bi = 0; bi < 4; ++bi) {
      int col = n0 + wc * 64 + bi * 16 + lm;
      float bv = (bias && col < bias_n) ? bias[col] : 0.0f;
#pragma unroll
      for (int r = 0; r < 8; ++r) {
        int row = m0 + wr * 32 + ai * 16 + r + 8 * half;
        float v = acc[ai][bi][r] + bv;
        if (act == 1) v = fmaxf(v, 0.0f);
        else if (act == 2) v = (v > 0.0f) ? v : v * SLOPE;
        if (Cf) {
          if (row < M && col < Nc) Cf[(long long)row * ldc + col] = v;
        } else {
          if (col < Nc) Cb[(long long)row * ldc + col] = f2bf(v);
        }
      }
    }
  }
}

// ---------------------------------------------------------------------------
// f32 [M x K] -> zero-padded bf16 [Mp x Kp]
// ---------------------------------------------------------------------------
__global__ __launch_bounds__(256)
void cvt_x_bf16(const float* __restrict__ in, bf16_t* __restrict__ out,
                int M, int K, int Mp, int Kp)
{
  long long idx = (long long)blockIdx.x * 256 + threadIdx.x;
  long long total = (long long)Mp * Kp;
  if (idx >= total) return;
  int r = (int)(idx / Kp), c = (int)(idx % Kp);
  float v = (r < M && c < K) ? in[(long long)r * K + c] : 0.0f;
  out[idx] = f2bf(v);
}

// f32 [K x Nc] -> transposed zero-padded bf16 [Ncp x Kp] (K-major per column)
__global__ __launch_bounds__(256)
void cvt_w_t_bf16(const float* __restrict__ in, bf16_t* __restrict__ out,
                  int K, int Nc, int Ncp, int Kp)
{
  long long idx = (long long)blockIdx.x * 256 + threadIdx.x;
  long long total = (long long)Ncp * Kp;
  if (idx >= total) return;
  int n = (int)(idx / Kp), k = (int)(idx % Kp);
  float v = (n < Nc && k < K) ? in[(long long)k * Nc + n] : 0.0f;
  out[idx] = f2bf(v);
}

// ---------------------------------------------------------------------------
// Edge attention logits: logit[e] = sum_c q[dst,c]*k[src,c]. One wave / edge.
// ---------------------------------------------------------------------------
__global__ __launch_bounds__(256)
void edge_dot(const float* __restrict__ q, const float* __restrict__ k,
              const long long* __restrict__ ei, float* __restrict__ logit)
{
  int e = blockIdx.x * 8 + (threadIdx.x >> 5);
  int lane = threadIdx.x & 31;
  if (e >= E_EDGES) return;
  int s = (int)ei[e];
  int d = (int)ei[E_EDGES + e];
  float4 qv = *(const float4*)&q[(long long)d * DH + lane * 4];
  float4 kv = *(const float4*)&k[(long long)s * DH + lane * 4];
  float sum = qv.x * kv.x + qv.y * kv.y + qv.z * kv.z + qv.w * kv.w;
#pragma unroll
  for (int off = 16; off > 0; off >>= 1) sum += __shfl_xor(sum, off, 32);
  if (lane == 0) logit[e] = sum;
}

// ---------------------------------------------------------------------------
// Global softmax reductions (two-level, deterministic).
// ---------------------------------------------------------------------------
#define RBLK 1024

__global__ __launch_bounds__(256)
void reduce_max_part(const float* __restrict__ in, int n, float* __restrict__ part)
{
  __shared__ float sm[256];
  float m = -3.4e38f;
  for (long long i = (long long)blockIdx.x * 256 + threadIdx.x; i < n;
       i += (long long)gridDim.x * 256)
    m = fmaxf(m, in[i]);
  sm[threadIdx.x] = m; __syncthreads();
  for (int s = 128; s > 0; s >>= 1) {
    if (threadIdx.x < s) sm[threadIdx.x] = fmaxf(sm[threadIdx.x], sm[threadIdx.x + s]);
    __syncthreads();
  }
  if (threadIdx.x == 0) part[blockIdx.x] = sm[0];
}

__global__ __launch_bounds__(256)
void reduce_max_final(const float* __restrict__ part, float* __restrict__ outv)
{
  __shared__ float sm[256];
  int t = threadIdx.x;
  float m = fmaxf(fmaxf(part[t], part[t + 256]), fmaxf(part[t + 512], part[t + 768]));
  sm[t] = m; __syncthreads();
  for (int s = 128; s > 0; s >>= 1) {
    if (t < s) sm[t] = fmaxf(sm[t], sm[t + s]);
    __syncthreads();
  }
  if (t == 0) outv[0] = sm[0];
}

__global__ __launch_bounds__(256)
void reduce_sumexp_part(const float* __restrict__ in, int n,
                        const float* __restrict__ gmaxp, float* __restrict__ part)
{
  __shared__ float sm[256];
  float gmax = gmaxp[0];
  float acc = 0.0f;
  for (long long i = (long long)blockIdx.x * 256 + threadIdx.x; i < n;
       i += (long long)gridDim.x * 256)
    acc += __expf(in[i] - gmax);
  sm[threadIdx.x] = acc; __syncthreads();
  for (int s = 128; s > 0; s >>= 1) {
    if (threadIdx.x < s) sm[threadIdx.x] += sm[threadIdx.x + s];
    __syncthreads();
  }
  if (threadIdx.x == 0) part[blockIdx.x] = sm[0];
}

__global__ __launch_bounds__(256)
void reduce_sum_final(const float* __restrict__ part, float* __restrict__ outv)
{
  __shared__ float sm[256];
  int t = threadIdx.x;
  float a = part[t] + part[t + 256] + part[t + 512] + part[t + 768];
  sm[t] = a; __syncthreads();
  for (int s = 128; s > 0; s >>= 1) {
    if (t < s) sm[t] += sm[t + s];
    __syncthreads();
  }
  if (t == 0) outv[0] = sm[0];
}

__global__ __launch_bounds__(256)
void fill_zero(float* __restrict__ p, long long n)
{
  long long i = (long long)blockIdx.x * 256 + threadIdx.x;
  if (i < n) p[i] = 0.0f;
}

// ---------------------------------------------------------------------------
// aggr[dst] += softmax(logit)[e] * v[src] * sigmoid(ea + hi[src] + hj[dst])
// One thread per (edge, 4 channels).
// ---------------------------------------------------------------------------
__global__ __launch_bounds__(256)
void edge_scatter(const float* __restrict__ logit, const float* __restrict__ gmaxp,
                  const float* __restrict__ gsump,
                  const float* __restrict__ v, const float* __restrict__ hi,
                  const float* __restrict__ hj, const float* __restrict__ ea,
                  const long long* __restrict__ ei, float* __restrict__ aggr)
{
  long long idx = (long long)blockIdx.x * 256 + threadIdx.x;
  int e = (int)(idx >> 5);
  if (e >= E_EDGES) return;
  int c = ((int)idx & 31) * 4;
  int s = (int)ei[e];
  int d = (int)ei[E_EDGES + e];
  float w = __expf(logit[e] - gmaxp[0]) / gsump[0];
  float4 g4  = *(const float4*)&ea[(long long)e * DH + c];
  float4 hi4 = *(const float4*)&hi[(long long)s * DH + c];
  float4 hj4 = *(const float4*)&hj[(long long)d * DH + c];
  float4 v4  = *(const float4*)&v [(long long)s * DH + c];
  float* ad = &aggr[(long long)d * DH + c];
  float m0 = w * v4.x / (1.0f + __expf(-(g4.x + hi4.x + hj4.x)));
  float m1 = w * v4.y / (1.0f + __expf(-(g4.y + hi4.y + hj4.y)));
  float m2 = w * v4.z / (1.0f + __expf(-(g4.z + hi4.z + hj4.z)));
  float m3 = w * v4.w / (1.0f + __expf(-(g4.w + hi4.w + hj4.w)));
  atomicAdd(&ad[0], m0);
  atomicAdd(&ad[1], m1);
  atomicAdd(&ad[2], m2);
  atomicAdd(&ad[3], m3);
}

// ---------------------------------------------------------------------------
// out = LayerNorm(a + b) * g + be, row-wise over DH=128. One wave per row.
// ---------------------------------------------------------------------------
__global__ __launch_bounds__(256)
void add_layernorm(const float* __restrict__ a, const float* __restrict__ b,
                   const float* __restrict__ g, const float* __restrict__ be,
                   float* __restrict__ out, int n)
{
  int row = blockIdx.x * 8 + (threadIdx.x >> 5);
  int lane = threadIdx.x & 31;
  if (row >= n) return;
  long long base = (long long)row * DH + lane * 4;
  float4 va = *(const float4*)&a[base];
  float4 vb = *(const float4*)&b[base];
  float x0 = va.x + vb.x, x1 = va.y + vb.y, x2 = va.z + vb.z, x3 = va.w + vb.w;
  float s = x0 + x1 + x2 + x3;
#pragma unroll
  for (int off = 16; off > 0; off >>= 1) s += __shfl_xor(s, off, 32);
  float mean = s * (1.0f / DH);
  float d0 = x0 - mean, d1 = x1 - mean, d2 = x2 - mean, d3 = x3 - mean;
  float vs = d0 * d0 + d1 * d1 + d2 * d2 + d3 * d3;
#pragma unroll
  for (int off = 16; off > 0; off >>= 1) vs += __shfl_xor(vs, off, 32);
  float rstd = rsqrtf(vs * (1.0f / DH) + EPS_LN);
  int c = lane * 4;
  out[base + 0] = d0 * rstd * g[c + 0] + be[c + 0];
  out[base + 1] = d1 * rstd * g[c + 1] + be[c + 1];
  out[base + 2] = d2 * rstd * g[c + 2] + be[c + 2];
  out[base + 3] = d3 * rstd * g[c + 3] + be[c + 3];
}

// ---------------------------------------------------------------------------
extern "C" void kernel_launch(void* const* d_in, const int* in_sizes, int n_in,
                              void* d_out, int out_size, void* d_ws, size_t ws_size,
                              hipStream_t stream)
{
  const float*     x_in  = (const float*)d_in[0];
  const long long* ei    = (const long long*)d_in[1];
  const float*     ea    = (const float*)d_in[2];
  const float* Wq  = (const float*)d_in[3];  const float* bq  = (const float*)d_in[4];
  const float* Wk  = (const float*)d_in[5];  const float* bk  = (const float*)d_in[6];
  const float* Wv  = (const float*)d_in[7];  const float* bv  = (const float*)d_in[8];
  const float* Wr  = (const float*)d_in[9];  const float* br  = (const float*)d_in[10];
  const float* Whi = (const float*)d_in[11]; const float* Whj = (const float*)d_in[12];
  const float* W1  = (const float*)d_in[13]; const float* b1  = (const float*)d_in[14];
  const float* W2  = (const float*)d_in[15]; const float* b2  = (const float*)d_in[16];
  const float* g1  = (const float*)d_in[17]; const float* be1 = (const float*)d_in[18];
  const float* g2  = (const float*)d_in[19]; const float* be2 = (const float*)d_in[20];
  const float* linW  = (const float*)d_in[21]; const float* linb  = (const float*)d_in[22];
  const float* lin2W = (const float*)d_in[23]; const float* lin2b = (const float*)d_in[24];

  char* wsb = (char*)d_ws;
  size_t off = 0;
  auto carve = [&](size_t bytes) -> void* {
    void* p = wsb + off;
    off += (bytes + 255) & ~(size_t)255;
    return p;
  };

  // bf16 activation buffers (padded)
  bf16_t* xbf0  = (bf16_t*)carve((size_t)MP * KP_IN * 2);
  bf16_t* xbf1  = (bf16_t*)carve((size_t)MP * KP_IN * 2);
  bf16_t* ssbf  = (bf16_t*)carve((size_t)MP * DH * 2);
  bf16_t* hbbf  = (bf16_t*)carve((size_t)MP * DH * 2);
  bf16_t* ff1bf = (bf16_t*)carve((size_t)MP * DFF * 2);
  // bf16 transposed weights
  bf16_t* Wqt[L_LAYERS], *Wkt[L_LAYERS], *Wvt[L_LAYERS], *Wrt[L_LAYERS];
  bf16_t* Whit[L_LAYERS], *Whjt[L_LAYERS], *W1t[L_LAYERS], *W2t[L_LAYERS];
  for (int l = 0; l < L_LAYERS; ++l) {
    Wqt[l]  = (bf16_t*)carve((size_t)DH * KP_IN * 2);
    Wkt[l]  = (bf16_t*)carve((size_t)DH * KP_IN * 2);
    Wvt[l]  = (bf16_t*)carve((size_t)DH * KP_IN * 2);
    Wrt[l]  = (bf16_t*)carve((size_t)DH * KP_IN * 2);
    Whit[l] = (bf16_t*)carve((size_t)DH * KP_IN * 2);
    Whjt[l] = (bf16_t*)carve((size_t)DH * KP_IN * 2);
    W1t[l]  = (bf16_t*)carve((size_t)DFF * DH * 2);
    W2t[l]  = (bf16_t*)carve((size_t)DH * DFF * 2);
  }
  bf16_t* linWt  = (bf16_t*)carve((size_t)NCP_LIN * DH * 2);
  bf16_t* lin2Wt = (bf16_t*)carve((size_t)DH * KP_IN * 2);
  // f32 buffers
  float* qb    = (float*)carve((size_t)N_NODES * DH * 4);
  float* kb    = (float*)carve((size_t)N_NODES * DH * 4);
  float* vb    = (float*)carve((size_t)N_NODES * DH * 4);
  float* rb    = (float*)carve((size_t)N_NODES * DH * 4);
  float* hib   = (float*)carve((size_t)N_NODES * DH * 4);
  float* hjb   = (float*)carve((size_t)N_NODES * DH * 4);
  float* logit = (float*)carve((size_t)E_EDGES * 4);
  float* aggr  = (float*)carve((size_t)N_NODES * DH * 4);
  float* ssb   = (float*)carve((size_t)N_NODES * DH * 4);
  float* ss2   = (float*)carve((size_t)N_NODES * DH * 4);
  float* hb    = (float*)carve((size_t)N_NODES * DH * 4);
  float* part  = (float*)carve(RBLK * 4);
  float* stats = (float*)carve(4 * 4);

  dim3 blk(256);
  const int gm = MP / BM;  // 157
  auto cblocks = [](long long n) { return (int)((n + 255) / 256); };

  // ---- one-time (per launch) conversions: x and all weights -> bf16 ----
  cvt_x_bf16<<<cblocks((long long)MP * KP_IN), blk, 0, stream>>>(x_in, xbf0, N_NODES, DIN, MP, KP_IN);
  for (int l = 0; l < L_LAYERS; ++l) {
    size_t wo = (size_t)l * DIN * DH;
    cvt_w_t_bf16<<<cblocks((long long)DH * KP_IN), blk, 0, stream>>>(Wq  + wo, Wqt[l],  DIN, DH, DH, KP_IN);
    cvt_w_t_bf16<<<cblocks((long long)DH * KP_IN), blk, 0, stream>>>(Wk  + wo, Wkt[l],  DIN, DH, DH, KP_IN);
    cvt_w_t_bf16<<<cblocks((long long)DH * KP_IN), blk, 0, stream>>>(Wv  + wo, Wvt[l],  DIN, DH, DH, KP_IN);
    cvt_w_t_bf16<<<cblocks((long long)DH * KP_IN), blk, 0, stream>>>(Wr  + wo, Wrt[l],  DIN, DH, DH, KP_IN);
    cvt_w_t_bf16<<<cblocks((long long)DH * KP_IN), blk, 0, stream>>>(Whi + wo, Whit[l], DIN, DH, DH, KP_IN);
    cvt_w_t_bf16<<<cblocks((long long)DH * KP_IN), blk, 0, stream>>>(Whj + wo, Whjt[l], DIN, DH, DH, KP_IN);
    cvt_w_t_bf16<<<cblocks((long long)DFF * DH), blk, 0, stream>>>(W1 + (size_t)l * DH * DFF, W1t[l], DH, DFF, DFF, DH);
    cvt_w_t_bf16<<<cblocks((long long)DH * DFF), blk, 0, stream>>>(W2 + (size_t)l * DFF * DH, W2t[l], DFF, DH, DH, DFF);
  }
  cvt_w_t_bf16<<<cblocks((long long)NCP_LIN * DH), blk, 0, stream>>>(linW,  linWt,  DH, DIN, NCP_LIN, DH);
  cvt_w_t_bf16<<<cblocks((long long)DH * KP_IN), blk, 0, stream>>>(lin2W, lin2Wt, DIN, DH, DH, KP_IN);

  bf16_t* xcur = xbf0;
  bf16_t* xnxt = xbf1;

  for (int l = 0; l < L_LAYERS; ++l) {
    size_t bo = (size_t)l * DH;

    // --- six projections from x (Kp = 1568), f32 out [N x 128] ---
    gemm_bf16_wmma<<<dim3(gm, 1), blk, 0, stream>>>(xcur, KP_IN, Wqt[l],  KP_IN, bq + bo, DH, 0, qb,  nullptr, DH, N_NODES, DH, KP_IN);
    gemm_bf16_wmma<<<dim3(gm, 1), blk, 0, stream>>>(xcur, KP_IN, Wkt[l],  KP_IN, bk + bo, DH, 0, kb,  nullptr, DH, N_NODES, DH, KP_IN);
    gemm_bf16_wmma<<<dim3(gm, 1), blk, 0, stream>>>(xcur, KP_IN, Wvt[l],  KP_IN, bv + bo, DH, 0, vb,  nullptr, DH, N_NODES, DH, KP_IN);
    gemm_bf16_wmma<<<dim3(gm, 1), blk, 0, stream>>>(xcur, KP_IN, Wrt[l],  KP_IN, br + bo, DH, 0, rb,  nullptr, DH, N_NODES, DH, KP_IN);
    gemm_bf16_wmma<<<dim3(gm, 1), blk, 0, stream>>>(xcur, KP_IN, Whit[l], KP_IN, nullptr, 0,  0, hib, nullptr, DH, N_NODES, DH, KP_IN);
    gemm_bf16_wmma<<<dim3(gm, 1), blk, 0, stream>>>(xcur, KP_IN, Whjt[l], KP_IN, nullptr, 0,  0, hjb, nullptr, DH, N_NODES, DH, KP_IN);

    // --- edge attention: logits, global softmax, gated scatter-add ---
    edge_dot<<<E_EDGES / 8, blk, 0, stream>>>(qb, kb, ei, logit);
    reduce_max_part<<<RBLK, blk, 0, stream>>>(logit, E_EDGES, part);
    reduce_max_final<<<1, blk, 0, stream>>>(part, stats + 0);
    reduce_sumexp_part<<<RBLK, blk, 0, stream>>>(logit, E_EDGES, stats + 0, part);
    reduce_sum_final<<<1, blk, 0, stream>>>(part, stats + 1);
    fill_zero<<<cblocks((long long)N_NODES * DH), blk, 0, stream>>>(aggr, (long long)N_NODES * DH);
    edge_scatter<<<cblocks((long long)E_EDGES * 32), blk, 0, stream>>>(
        logit, stats + 0, stats + 1, vb, hib, hjb, ea, ei, aggr);

    // --- FF MLP with post-norm residual ---
    add_layernorm<<<(N_NODES + 7) / 8, blk, 0, stream>>>(aggr, rb, g1 + bo, be1 + bo, ssb, N_NODES);
    cvt_x_bf16<<<cblocks((long long)MP * DH), blk, 0, stream>>>(ssb, ssbf, N_NODES, DH, MP, DH);
    gemm_bf16_wmma<<<dim3(gm, DFF / BN), blk, 0, stream>>>(ssbf, DH, W1t[l], DH,
        b1 + (size_t)l * DFF, DFF, 1, nullptr, ff1bf, DFF, N_NODES, DFF, DH);
    gemm_bf16_wmma<<<dim3(gm, 1), blk, 0, stream>>>(ff1bf, DFF, W2t[l], DFF,
        b2 + bo, DH, 0, ss2, nullptr, DH, N_NODES, DH, DFF);
    add_layernorm<<<(N_NODES + 7) / 8, blk, 0, stream>>>(ssb, ss2, g2 + bo, be2 + bo, hb, N_NODES);
    cvt_x_bf16<<<cblocks((long long)MP * DH), blk, 0, stream>>>(hb, hbbf, N_NODES, DH, MP, DH);

    // --- shared projection 128 -> 1546, direct bf16 padded output ---
    gemm_bf16_wmma<<<dim3(gm, (KP_IN + BN - 1) / BN), blk, 0, stream>>>(hbbf, DH, linWt, DH,
        linb, DIN, 0, nullptr, xnxt, KP_IN, N_NODES, KP_IN, DH);

    bf16_t* t = xcur; xcur = xnxt; xnxt = t;
  }

  // --- final linear 1546 -> 128 + leaky ReLU -> d_out (f32) ---
  gemm_bf16_wmma<<<dim3(gm, 1), blk, 0, stream>>>(xcur, KP_IN, lin2Wt, KP_IN,
      lin2b, DH, 2, (float*)d_out, nullptr, DH, N_NODES, DH, KP_IN);
}
